// BinaryLinear_48601849921988
// MI455X (gfx1250) — matrix operand order repaired and checked by reading, not compiled, not run on hardware
//
#include <hip/hip_runtime.h>

// ---------------------------------------------------------------------------
// Binary linear: out = x @ sign(w)^T + b
//   x: [8192, 2048] f32   w: [4096, 2048] f32   b: [4096] f32
//   out: [8192, 4096] f32
// bf16 WMMA path (v_wmma_f32_16x16x32_bf16, f32 accumulate):
//   - sign(w) in {-1,0,1} is exact in bf16
//   - x -> bf16 RNE (hardware v_cvt_pk_bf16_f32 when available)
// 128x128 C tile per 256-thread block (8 wave32, 2x4 wave grid, 64x32/wave),
// BK=32 double-buffered LDS (40 KB, padded rows -> conflict-free b128 loads).
// ---------------------------------------------------------------------------

typedef __bf16 v2bf  __attribute__((ext_vector_type(2)));
typedef __bf16 v8bf  __attribute__((ext_vector_type(8)));
typedef __bf16 v16bf __attribute__((ext_vector_type(16)));
typedef float  v8f   __attribute__((ext_vector_type(8)));

#define M_DIM 8192
#define N_DIM 4096
#define K_DIM 2048
#define BM 128
#define BN 128
#define BK 32
#define LSTR 40   // 32 bf16 + 8 pad -> 80B row stride, conflict-free b128 frag loads

__device__ __forceinline__ unsigned int bf16pk_rne(float lo, float hi) {
#if __has_builtin(__builtin_amdgcn_cvt_pk_bf16_f32)
    v2bf p = __builtin_amdgcn_cvt_pk_bf16_f32(lo, hi);
    union { v2bf v; unsigned int u; } cvt;
    cvt.v = p;
    return cvt.u;
#else
    unsigned int a = __float_as_uint(lo);
    unsigned int b = __float_as_uint(hi);
    a = (a + 0x7FFFu + ((a >> 16) & 1u)) >> 16;
    b = (b + 0x7FFFu + ((b >> 16) & 1u)) >> 16;
    return a | (b << 16);
#endif
}

// sign(f) as bf16 bits via integer ops only (no float compares):
//   0 if f == +/-0, else sign-bit | 0x3F80  (= +/-1.0bf16)
__device__ __forceinline__ unsigned int sign_bf16_bits(float f) {
    const unsigned int u   = __float_as_uint(f);
    const unsigned int mag = u & 0x7FFFFFFFu;
    const unsigned int s   = ((u >> 16) & 0x8000u) | 0x3F80u;
    return mag ? s : 0u;
}

__device__ __forceinline__ unsigned int signpk_bf16(float lo, float hi) {
    return sign_bf16_bits(lo) | (sign_bf16_bits(hi) << 16);
}

__global__ __launch_bounds__(256)
void binlin_bf16_wmma(const float* __restrict__ X,
                      const float* __restrict__ W,
                      const float* __restrict__ Bias,
                      float* __restrict__ Out) {
    __shared__ alignas(16) unsigned short As[2][BM][LSTR];
    __shared__ alignas(16) unsigned short Bs[2][BN][LSTR];

    const int tid      = threadIdx.x;
    const int lane     = tid & 31;
    const int wave     = tid >> 5;
    const int laneMod  = lane & 15;
    const int laneHalf = lane >> 4;
    const int waveM    = wave >> 2;   // 0..1 -> 64-row slab
    const int waveN    = wave & 3;    // 0..3 -> 32-col slab

    const int blockN = blockIdx.x * BN;
    const int blockM = blockIdx.y * BM;

    // staging map: 128 rows x 32 K-floats per tile = 1024 float4 / 256 thr = 4 each
    const int lcol = (tid & 7) * 4;   // float column within K-block
    const int lrow = tid >> 3;        // 0..31, stride 32

    v8f acc[4][2];
#pragma unroll
    for (int mt = 0; mt < 4; ++mt)
#pragma unroll
        for (int nt = 0; nt < 2; ++nt)
            acc[mt][nt] = (v8f){0.f, 0.f, 0.f, 0.f, 0.f, 0.f, 0.f, 0.f};

    auto stage = [&](int k0, int buf) {
#pragma unroll
        for (int i = 0; i < 4; ++i) {
            const int r = lrow + i * 32;
            const float4 xa = *reinterpret_cast<const float4*>(
                X + (size_t)(blockM + r) * K_DIM + (k0 + lcol));
            const unsigned int p0 = bf16pk_rne(xa.x, xa.y);
            const unsigned int p1 = bf16pk_rne(xa.z, xa.w);
            *reinterpret_cast<uint2*>(&As[buf][r][lcol]) = make_uint2(p0, p1);

            const float4 wa = *reinterpret_cast<const float4*>(
                W + (size_t)(blockN + r) * K_DIM + (k0 + lcol));
            const unsigned int q0 = signpk_bf16(wa.x, wa.y);
            const unsigned int q1 = signpk_bf16(wa.z, wa.w);
            *reinterpret_cast<uint2*>(&Bs[buf][r][lcol]) = make_uint2(q0, q1);
        }
    };

    stage(0, 0);
    __syncthreads();

    int buf = 0;
    const int NKB = K_DIM / BK;   // 64
    for (int kb = 0; kb < NKB; ++kb) {
        if (kb + 1 < NKB) stage((kb + 1) * BK, buf ^ 1);

        // --- fragment loads from LDS (contiguous k = 16*laneHalf + s, same for A and B) ---
        v16bf afrag[4];
        v16bf bfrag[2];
#pragma unroll
        for (int nt = 0; nt < 2; ++nt) {
            const int c = waveN * 32 + nt * 16 + laneMod;
            const v8bf lo = *reinterpret_cast<const v8bf*>(&Bs[buf][c][laneHalf * 16]);
            const v8bf hi = *reinterpret_cast<const v8bf*>(&Bs[buf][c][laneHalf * 16 + 8]);
            bfrag[nt] = __builtin_shufflevector(lo, hi,
                0, 1, 2, 3, 4, 5, 6, 7, 8, 9, 10, 11, 12, 13, 14, 15);
        }
#pragma unroll
        for (int mt = 0; mt < 4; ++mt) {
            const int r = waveM * 64 + mt * 16 + laneMod;
            const v8bf lo = *reinterpret_cast<const v8bf*>(&As[buf][r][laneHalf * 16]);
            const v8bf hi = *reinterpret_cast<const v8bf*>(&As[buf][r][laneHalf * 16 + 8]);
            afrag[mt] = __builtin_shufflevector(lo, hi,
                0, 1, 2, 3, 4, 5, 6, 7, 8, 9, 10, 11, 12, 13, 14, 15);
        }

        // --- 8 WMMAs per wave per K-step ---
#pragma unroll
        for (int mt = 0; mt < 4; ++mt)
#pragma unroll
            for (int nt = 0; nt < 2; ++nt)
                acc[mt][nt] = __builtin_amdgcn_wmma_f32_16x16x32_bf16(
                    false, afrag[mt], false, bfrag[nt],
                    (short)0, acc[mt][nt], false, false);

        __syncthreads();
        buf ^= 1;
    }

    // --- epilogue: bias add + f32 store (C layout: vgpr i -> M = i + 8*laneHalf,
    //     lane%16 -> N) ---
#pragma unroll
    for (int nt = 0; nt < 2; ++nt) {
        const int col  = blockN + waveN * 32 + nt * 16 + laneMod;
        const float bv = Bias[col];
#pragma unroll
        for (int mt = 0; mt < 4; ++mt) {
            const int rbase = blockM + waveM * 64 + mt * 16 + laneHalf * 8;
            float* __restrict__ orow = Out + (size_t)rbase * N_DIM + col;
#pragma unroll
            for (int i = 0; i < 8; ++i) {
                orow[(size_t)i * N_DIM] = acc[mt][nt][i] + bv;
            }
        }
    }
}

extern "C" void kernel_launch(void* const* d_in, const int* in_sizes, int n_in,
                              void* d_out, int out_size, void* d_ws, size_t ws_size,
                              hipStream_t stream) {
    (void)in_sizes; (void)n_in; (void)d_ws; (void)ws_size; (void)out_size;
    const float* x = (const float*)d_in[0];
    const float* w = (const float*)d_in[1];
    const float* b = (const float*)d_in[2];
    float* out = (float*)d_out;

    dim3 grid(N_DIM / BN, M_DIM / BM);   // (32, 64)
    dim3 block(256);
    binlin_bf16_wmma<<<grid, block, 0, stream>>>(x, w, b, out);
}
